// MultiHeadAttention_32315333935918
// MI455X (gfx1250) — compile-verified, matrix operand order checked by reading, not run on hardware
//
#include <hip/hip_runtime.h>
#include <hip/hip_bf16.h>

typedef __attribute__((ext_vector_type(16))) __bf16 v16bf;
typedef __attribute__((ext_vector_type(8)))  float  v8f;
typedef __attribute__((ext_vector_type(4)))  unsigned int v4u;
typedef __attribute__((ext_vector_type(8)))  int v8i;
typedef __attribute__((ext_vector_type(4)))  int v4i;

// ---------------------------------------------------------------------------
// WMMA helpers (CDNA5 gfx1250, wave32). D = A(16x32 bf16) * B(32x16 bf16) + C
// ---------------------------------------------------------------------------
__device__ __forceinline__ v8f wmma_bf16(v16bf a, v16bf b, v8f c) {
  return __builtin_amdgcn_wmma_f32_16x16x32_bf16(
      /*neg_a=*/false, a, /*neg_b=*/false, b,
      /*c_mod=*/(short)0, c, /*reuse_a=*/false, /*reuse_b=*/false);
}

// A fragment: 16x32 tile, row-major, leading dim `ld`.
// lanes 0-15: row M=lane, K = 0..7 & 16..23 ; lanes 16-31: K = 8..15 & 24..31
__device__ __forceinline__ v16bf load_a_frag(const __bf16* __restrict__ base, int ld) {
  const int lane = threadIdx.x & 31;
  const int half = lane >> 4;
  const int m    = lane & 15;
  v16bf a;
#pragma unroll
  for (int i = 0; i < 16; ++i) {
    int k = (i & 7) + ((i >> 3) << 4) + (half << 3);
    a[i] = base[m * ld + k];
  }
  return a;
}

// B fragment: 32x16 tile (K x N); element (k,n) at base[n*ldn + k*ldk].
__device__ __forceinline__ v16bf load_b_frag(const __bf16* __restrict__ base,
                                             int ldk, int ldn) {
  const int lane = threadIdx.x & 31;
  const int half = lane >> 4;
  const int n    = lane & 15;
  v16bf b;
#pragma unroll
  for (int i = 0; i < 16; ++i) {
    int k = (i & 7) + ((i >> 3) << 4) + (half << 3);
    b[i] = base[n * ldn + k * ldk];
  }
  return b;
}

// ---------------------------------------------------------------------------
// TDM: DMA a 2-D bf16 tile (tile_d1 rows x tile_d0 cols, row-major tensor
// with line stride `stride0` elements) from global into LDS at `lds_off`.
// D# per CDNA5 ISA ch.8: group0 = {count|flags, lds_addr, global_addr, type},
// group1 = {mask/data_size, dims, tile dims, strides}; groups 2/3 zero (2-D).
// ---------------------------------------------------------------------------
__device__ __forceinline__ void tdm_load_2d(unsigned lds_off, const void* gptr,
                                            unsigned tensor_d0, unsigned tensor_d1,
                                            unsigned tile_d0, unsigned tile_d1,
                                            unsigned stride0) {
  unsigned long long ga = (unsigned long long)(size_t)gptr;
  v4u g0;
  g0[0] = 1u;                                            // count=1, user mode
  g0[1] = lds_off;                                       // lds_addr (bytes)
  g0[2] = (unsigned)(ga & 0xffffffffu);                  // global_addr[31:0]
  g0[3] = (unsigned)((ga >> 32) & 0x01ffffffu) | (2u << 30);  // [56:32] | type=2
  v8i g1;
  g1[0] = (int)(1u << 16);                               // data_size=1 (2 bytes)
  g1[1] = (int)((tensor_d0 & 0xffffu) << 16);            // tensor_dim0[15:0]
  g1[2] = (int)(((tensor_d0 >> 16) & 0xffffu) |
                ((tensor_d1 & 0xffffu) << 16));          // dim0 hi | dim1 lo
  g1[3] = (int)(((tensor_d1 >> 16) & 0xffffu) |
                ((tile_d0 & 0xffffu) << 16));            // dim1 hi | tile_dim0
  g1[4] = (int)(tile_d1 & 0xffffu);                      // tile_dim1 (tile_dim2=0)
  g1[5] = (int)stride0;                                  // tensor_dim0_stride lo32
  g1[6] = 0;                                             // stride0 hi | dim1_stride lo
  g1[7] = 0;
  v4i z4 = {0, 0, 0, 0};
#if defined(__clang_major__) && (__clang_major__ >= 23)
  v8i z8 = {0, 0, 0, 0, 0, 0, 0, 0};
  __builtin_amdgcn_tensor_load_to_lds(g0, g1, z4, z4, z8, 0);
#else
  __builtin_amdgcn_tensor_load_to_lds(g0, g1, z4, z4, 0);
#endif
}

// ---------------------------------------------------------------------------
// Elementwise f32 -> bf16 cast
// ---------------------------------------------------------------------------
__global__ void cast_bf16_kernel(const float* __restrict__ src,
                                 __bf16* __restrict__ dst, int n) {
  int i = blockIdx.x * blockDim.x + threadIdx.x;
  if (i < n) dst[i] = (__bf16)src[i];
}

// ---------------------------------------------------------------------------
// Fused QKV projection: X[8192,1024] * Win^T[1024,3072] + b_in.
// Wave computes a 16x64 tile (A-frag reused across 4 WMMAs per k-step).
// Scatters Q (pre-scaled 1/8) and K as [B,H,S,hd]; V transposed [B,H,hd,S].
// grid = (8192/16, 3072/256), block = 128 (4 waves)
// ---------------------------------------------------------------------------
__global__ void __launch_bounds__(128)
qkv_gemm_kernel(const __bf16* __restrict__ X, const __bf16* __restrict__ Win,
                const float* __restrict__ b_in,
                __bf16* __restrict__ Q, __bf16* __restrict__ K,
                __bf16* __restrict__ Vt) {
  const int lane = threadIdx.x & 31;
  const int w    = threadIdx.x >> 5;
  const int half = lane >> 4;
  const int n    = lane & 15;
  const int row0 = blockIdx.x * 16;
  const int col0 = (blockIdx.y * 4 + w) * 64;

  v8f c0 = {}, c1 = {}, c2 = {}, c3 = {};
  for (int kk = 0; kk < 1024; kk += 32) {
    const __bf16* ap = X + (long)row0 * 1024 + kk;
    __builtin_prefetch(ap + 32, 0, 3);
    v16bf a = load_a_frag(ap, 1024);
    c0 = wmma_bf16(a, load_b_frag(Win + (long)(col0 +  0) * 1024 + kk, 1, 1024), c0);
    c1 = wmma_bf16(a, load_b_frag(Win + (long)(col0 + 16) * 1024 + kk, 1, 1024), c1);
    c2 = wmma_bf16(a, load_b_frag(Win + (long)(col0 + 32) * 1024 + kk, 1, 1024), c2);
    c3 = wmma_bf16(a, load_b_frag(Win + (long)(col0 + 48) * 1024 + kk, 1, 1024), c3);
  }

  v8f cs[4] = {c0, c1, c2, c3};
#pragma unroll
  for (int cfr = 0; cfr < 4; ++cfr) {
#pragma unroll
    for (int r = 0; r < 8; ++r) {
      int row = row0 + r + half * 8;   // b*S + s
      int e   = col0 + cfr * 16 + n;   // 0..3071
      float v = cs[cfr][r] + b_in[e];
      int bi = row >> 11;
      int s  = row & 2047;
      int h  = e / 192;
      int t  = (e % 192) / 64;         // 0=q 1=k 2=v
      int d  = e % 64;
      if (t == 0) {
        Q[(((long)(bi * 16 + h) * 2048) + s) * 64 + d] = (__bf16)(v * 0.125f);
      } else if (t == 1) {
        K[(((long)(bi * 16 + h) * 2048) + s) * 64 + d] = (__bf16)v;
      } else {
        Vt[((long)(bi * 16 + h) * 64 + d) * 2048 + s] = (__bf16)v;   // transposed
      }
    }
  }
}

// ---------------------------------------------------------------------------
// Flash attention. Block = 128 threads (4 waves) covering 128 queries of one
// (b,h); each wave owns 32 queries (two 16-row tiles).  K/V 32-key tiles are
// DMA'd once per block into LDS by the Tensor Data Mover (wave 0 issues,
// s_wait_tensorcnt drains, workgroup barrier publishes), then all 4 waves
// build WMMA fragments from LDS.  grid = (64, 16).
// ---------------------------------------------------------------------------
__global__ void __launch_bounds__(128)
attn_kernel(const __bf16* __restrict__ Q, const __bf16* __restrict__ K,
            const __bf16* __restrict__ Vt, __bf16* __restrict__ O) {
  __shared__ __bf16 Klds[32 * 64];         // K tile: [key][d]      (4 KB)
  __shared__ __bf16 Vlds[64 * 32];         // V tile: [d][key]      (4 KB)
  __shared__ __bf16 Plds[4][2][16 * 32];   // per-wave P staging    (8 KB)

  const int lane = threadIdx.x & 31;
  const int w    = threadIdx.x >> 5;
  const int half = lane >> 4;
  const int n    = lane & 15;
  const int bh   = blockIdx.x;                   // b*16 + h
  const int q0   = blockIdx.y * 128 + w * 32;

  const unsigned kls = (unsigned)(size_t)&Klds[0];
  const unsigned vls = (unsigned)(size_t)&Vlds[0];

  v16bf qa0[2], qa1[2];
#pragma unroll
  for (int qt = 0; qt < 2; ++qt) {
    const __bf16* Qb = Q + ((long)bh * 2048 + q0 + qt * 16) * 64;
    qa0[qt] = load_a_frag(Qb, 64);        // d = 0..31
    qa1[qt] = load_a_frag(Qb + 32, 64);   // d = 32..63
  }

  float mrow[2][8], lrow[2][8];
  v8f o[2][4];
#pragma unroll
  for (int qt = 0; qt < 2; ++qt) {
#pragma unroll
    for (int r = 0; r < 8; ++r) { mrow[qt][r] = -3.0e38f; lrow[qt][r] = 0.0f; }
#pragma unroll
    for (int cfr = 0; cfr < 4; ++cfr) o[qt][cfr] = (v8f){};
  }

  for (int j0 = 0; j0 < 2048; j0 += 32) {
    // --- TDM stage: K tile (32 keys x 64d) and V tile (64d x 32 keys) ------
    if (w == 0) {
      // K: tensor [2048,64] row-major, tile 32 rows x 64 cols at row j0
      tdm_load_2d(kls, K + ((long)bh * 2048 + j0) * 64,
                  /*tensor_d0=*/64, /*tensor_d1=*/2048,
                  /*tile_d0=*/64, /*tile_d1=*/32, /*stride0=*/64);
      // V: tensor [64,2048] row-major, tile 64 rows x 32 cols at col j0
      tdm_load_2d(vls, Vt + (long)bh * 64 * 2048 + j0,
                  /*tensor_d0=*/2048, /*tensor_d1=*/64,
                  /*tile_d0=*/32, /*tile_d1=*/64, /*stride0=*/2048);
      __builtin_amdgcn_s_wait_tensorcnt(0);
    }
    __syncthreads();

    // K^T fragments from LDS (element (k=d, n=key) = Klds[n*64 + k])
    v16bf kb00 = load_b_frag(Klds,                1, 64);
    v16bf kb01 = load_b_frag(Klds + 32,           1, 64);
    v16bf kb10 = load_b_frag(Klds + 16 * 64,      1, 64);
    v16bf kb11 = load_b_frag(Klds + 16 * 64 + 32, 1, 64);

#pragma unroll
    for (int qt = 0; qt < 2; ++qt) {
      v8f s0 = {}, s1 = {};
      s0 = wmma_bf16(qa0[qt], kb00, s0);
      s0 = wmma_bf16(qa1[qt], kb01, s0);
      s1 = wmma_bf16(qa0[qt], kb10, s1);
      s1 = wmma_bf16(qa1[qt], kb11, s1);

      // online softmax (VGPR r -> row r + half*8; lane&15 -> key column)
#pragma unroll
      for (int r = 0; r < 8; ++r) {
        float vmax = fmaxf(s0[r], s1[r]);
#pragma unroll
        for (int off = 8; off >= 1; off >>= 1)
          vmax = fmaxf(vmax, __shfl_xor(vmax, off, 16));
        float nm    = fmaxf(mrow[qt][r], vmax);
        float scale = __expf(mrow[qt][r] - nm);
        float p0    = __expf(s0[r] - nm);
        float p1    = __expf(s1[r] - nm);
        float rsum  = p0 + p1;
#pragma unroll
        for (int off = 8; off >= 1; off >>= 1)
          rsum += __shfl_xor(rsum, off, 16);
        lrow[qt][r] = lrow[qt][r] * scale + rsum;
        mrow[qt][r] = nm;
        o[qt][0][r] *= scale; o[qt][1][r] *= scale;
        o[qt][2][r] *= scale; o[qt][3][r] *= scale;
        Plds[w][qt][(r + half * 8) * 32 + n]      = (__bf16)p0;
        Plds[w][qt][(r + half * 8) * 32 + 16 + n] = (__bf16)p1;
      }
    }

    // V fragments from LDS (element (k=key, n=d) = Vlds[n*32 + k])
    v16bf vb0 = load_b_frag(Vlds +  0 * 32, 1, 32);
    v16bf vb1 = load_b_frag(Vlds + 16 * 32, 1, 32);
    v16bf vb2 = load_b_frag(Vlds + 32 * 32, 1, 32);
    v16bf vb3 = load_b_frag(Vlds + 48 * 32, 1, 32);

#pragma unroll
    for (int qt = 0; qt < 2; ++qt) {
      v16bf pa = load_a_frag(&Plds[w][qt][0], 32);
      o[qt][0] = wmma_bf16(pa, vb0, o[qt][0]);
      o[qt][1] = wmma_bf16(pa, vb1, o[qt][1]);
      o[qt][2] = wmma_bf16(pa, vb2, o[qt][2]);
      o[qt][3] = wmma_bf16(pa, vb3, o[qt][3]);
    }
    __syncthreads();   // protect LDS tiles before next TDM overwrite
  }

  const int bi = bh >> 4, h = bh & 15;
#pragma unroll
  for (int qt = 0; qt < 2; ++qt) {
#pragma unroll
    for (int r = 0; r < 8; ++r) {
      int s = q0 + qt * 16 + r + half * 8;
      float inv = 1.0f / lrow[qt][r];
      long base = ((long)bi * 2048 + s) * 1024 + h * 64 + n;
      O[base +  0] = (__bf16)(o[qt][0][r] * inv);
      O[base + 16] = (__bf16)(o[qt][1][r] * inv);
      O[base + 32] = (__bf16)(o[qt][2][r] * inv);
      O[base + 48] = (__bf16)(o[qt][3][r] * inv);
    }
  }
}

// ---------------------------------------------------------------------------
// Output projection: O[8192,1024] * Wout^T[1024,1024] + b_out -> fp32 d_out.
// Wave computes 16x64 tile. grid = (8192/16, 1024/256), block = 128
// ---------------------------------------------------------------------------
__global__ void __launch_bounds__(128)
out_gemm_kernel(const __bf16* __restrict__ O, const __bf16* __restrict__ Wout,
                const float* __restrict__ b_out, float* __restrict__ out) {
  const int lane = threadIdx.x & 31;
  const int w    = threadIdx.x >> 5;
  const int half = lane >> 4;
  const int n    = lane & 15;
  const int row0 = blockIdx.x * 16;
  const int col0 = (blockIdx.y * 4 + w) * 64;

  v8f c0 = {}, c1 = {}, c2 = {}, c3 = {};
  for (int kk = 0; kk < 1024; kk += 32) {
    const __bf16* ap = O + (long)row0 * 1024 + kk;
    __builtin_prefetch(ap + 32, 0, 3);
    v16bf a = load_a_frag(ap, 1024);
    c0 = wmma_bf16(a, load_b_frag(Wout + (long)(col0 +  0) * 1024 + kk, 1, 1024), c0);
    c1 = wmma_bf16(a, load_b_frag(Wout + (long)(col0 + 16) * 1024 + kk, 1, 1024), c1);
    c2 = wmma_bf16(a, load_b_frag(Wout + (long)(col0 + 32) * 1024 + kk, 1, 1024), c2);
    c3 = wmma_bf16(a, load_b_frag(Wout + (long)(col0 + 48) * 1024 + kk, 1, 1024), c3);
  }

  v8f cs[4] = {c0, c1, c2, c3};
#pragma unroll
  for (int cfr = 0; cfr < 4; ++cfr) {
#pragma unroll
    for (int r = 0; r < 8; ++r) {
      int row = row0 + r + half * 8;
      int f   = col0 + cfr * 16 + n;
      out[(long)row * 1024 + f] = cs[cfr][r] + b_out[f];
    }
  }
}

// ---------------------------------------------------------------------------
// Host side
// ---------------------------------------------------------------------------
extern "C" void kernel_launch(void* const* d_in, const int* in_sizes, int n_in,
                              void* d_out, int out_size, void* d_ws, size_t ws_size,
                              hipStream_t stream) {
  (void)in_sizes; (void)n_in; (void)out_size; (void)ws_size;
  const float* x     = (const float*)d_in[0];   // [4,2048,1024]
  const float* w_in  = (const float*)d_in[1];   // [3072,1024]
  const float* b_in  = (const float*)d_in[2];   // [3072]
  const float* w_out = (const float*)d_in[3];   // [1024,1024]
  const float* b_out = (const float*)d_in[4];   // [1024]
  float* out = (float*)d_out;                   // [4,2048,1024]

  const long nX  = 4L * 2048 * 1024;
  const long nWi = 3072L * 1024;
  const long nWo = 1024L * 1024;

  char* ws = (char*)d_ws;
  __bf16* Xbf    = (__bf16*)(ws);                       // 16 MB
  __bf16* WinBf  = (__bf16*)(ws + 16u * 1024 * 1024);   //  6 MB
  __bf16* WoutBf = (__bf16*)(ws + 22u * 1024 * 1024);   //  2 MB
  __bf16* Qbuf   = (__bf16*)(ws + 24u * 1024 * 1024);   // 16 MB
  __bf16* Kbuf   = (__bf16*)(ws + 40u * 1024 * 1024);   // 16 MB
  __bf16* Vtbuf  = (__bf16*)(ws + 56u * 1024 * 1024);   // 16 MB (transposed)
  __bf16* Obuf   = (__bf16*)(ws + 72u * 1024 * 1024);   // 16 MB

  cast_bf16_kernel<<<(nX  + 255) / 256, 256, 0, stream>>>(x,     Xbf,    (int)nX);
  cast_bf16_kernel<<<(nWi + 255) / 256, 256, 0, stream>>>(w_in,  WinBf,  (int)nWi);
  cast_bf16_kernel<<<(nWo + 255) / 256, 256, 0, stream>>>(w_out, WoutBf, (int)nWo);

  qkv_gemm_kernel<<<dim3(512, 12), 128, 0, stream>>>(Xbf, WinBf, b_in,
                                                     Qbuf, Kbuf, Vtbuf);
  attn_kernel<<<dim3(64, 16), 128, 0, stream>>>(Qbuf, Kbuf, Vtbuf, Obuf);
  out_gemm_kernel<<<dim3(512, 4), 128, 0, stream>>>(Obuf, WoutBf, b_out, out);
}
